// ImprovedGCN_37409165149044
// MI455X (gfx1250) — compile-verified, matrix operand order checked by reading, not run on hardware
//
#include <hip/hip_runtime.h>
#include <hip/hip_bf16.h>
#include <math.h>

// ---------------------------------------------------------------------------
// ImprovedGCN forward for MI455X (gfx1250, wave32, WMMA).
// Weight GEMMs: f16 pre-transposed B operands, 16x64 C-tile per wave
// (4 x v_wmma_f32_16x16x32_f16 per K-step, A-frag reused).
// ---------------------------------------------------------------------------

typedef __attribute__((ext_vector_type(16))) _Float16 v16h;
typedef __attribute__((ext_vector_type(8)))  float    v8f;

namespace gcn {
constexpr int Bn = 4, S = 128, D = 768, H = 8, NL = 3;
constexpr int DH = D / H;            // 96
constexpr int SD = S * D;            // 98304
constexpr int SS = S * S;            // 16384
constexpr int MR = Bn * S;           // 512 flattened rows
constexpr int D3 = 3 * D;            // 2304
constexpr int LC = S + 1;            // 129 conv output length
constexpr int DD = D * D;            // 589824

// workspace layout (floats)
constexpr size_t W_HI    = 0;
constexpr size_t W_HJ    = W_HI    + (size_t)MR * D;
constexpr size_t W_EW    = W_HJ    + (size_t)MR * D;
constexpr size_t W_ADJE  = W_EW    + (size_t)Bn * SS;
constexpr size_t W_DEN   = W_ADJE  + (size_t)Bn * SS;
constexpr size_t W_COL   = W_DEN   + MR;
constexpr size_t W_C129  = W_COL   + MR;                    // [Bn][D][129]
constexpr size_t W_MSI   = W_C129  + (size_t)Bn * D * LC;   // [Bn][3D][128]
constexpr size_t W_G     = W_MSI   + (size_t)Bn * D3 * S;   // [Bn][S][3D]
constexpr size_t W_FUSH  = W_G     + (size_t)Bn * S * D3;
constexpr size_t W_MSF   = W_FUSH  + (size_t)MR * D;
constexpr size_t W_Q     = W_MSF   + (size_t)MR * D;
constexpr size_t W_K     = W_Q     + (size_t)MR * D;
constexpr size_t W_V     = W_K     + (size_t)MR * D;
constexpr size_t W_ATT   = W_V     + (size_t)MR * D;        // [B*H][S][S]
constexpr size_t W_CTX   = W_ATT   + (size_t)Bn * H * SS;
constexpr size_t W_ATTO  = W_CTX   + (size_t)MR * D;
constexpr size_t W_ATTLN = W_ATTO  + (size_t)MR * D;
constexpr size_t W_TBUF  = W_ATTLN + (size_t)MR * D;
constexpr size_t W_RAXW  = W_TBUF  + (size_t)MR * D;
constexpr size_t W_PRELN = W_RAXW  + (size_t)MR * D;
constexpr size_t W_OUTA  = W_PRELN + (size_t)MR * D;
constexpr size_t W_OUTB  = W_OUTA  + (size_t)MR * D;
constexpr size_t W_CAT   = W_OUTB  + (size_t)MR * D;        // [512][1536]
constexpr size_t W_FEH   = W_CAT   + (size_t)MR * 2 * D;
constexpr size_t W_FE2   = W_FEH   + (size_t)MR * D;
constexpr size_t W_END   = W_FE2   + (size_t)MR * D;

// f16 pre-transposed weight region (element offsets within _Float16 region)
constexpr size_t F_EW1A = 0;                 // [768][768]
constexpr size_t F_EW1B = F_EW1A + DD;
constexpr size_t F_EW2  = F_EW1B + DD;
constexpr size_t F_WW   = F_EW2  + DD;       // 3 x [768][768]
constexpr size_t F_QW   = F_WW   + 3 * (size_t)DD;
constexpr size_t F_KW   = F_QW   + DD;
constexpr size_t F_VW   = F_KW   + DD;
constexpr size_t F_OW   = F_VW   + DD;
constexpr size_t F_FUS1 = F_OW   + DD;       // [768][2304]
constexpr size_t F_FUS2 = F_FUS1 + (size_t)D3 * D;
constexpr size_t F_FE1  = F_FUS2 + DD;       // [768][1536]
constexpr size_t F_FE2  = F_FE1  + (size_t)2 * D * D;
constexpr size_t F_END  = F_FE2  + DD;
} // namespace gcn

// --------------------------- WMMA fragment helpers -------------------------
// 16-bit A-matrix VGPR layout (ISA 7.12.2): lane l holds row m=l&15,
// element i (i<8) = K kh+i, element 8+i = K kh+16+i, with kh = 8*(l>=16).
__device__ __forceinline__ v16h frag_from_f32(const float* __restrict__ p0,
                                              const float* __restrict__ p1) {
  v16h f;
#pragma unroll
  for (int i = 0; i < 8; ++i) { f[i] = (_Float16)p0[i]; f[i + 8] = (_Float16)p1[i]; }
  return f;
}
__device__ __forceinline__ v16h frag_from_f32_strided(const float* __restrict__ base,
                                                      int stride) {
  v16h f;
#pragma unroll
  for (int i = 0; i < 8; ++i) {
    f[i]     = (_Float16)base[(size_t)i * stride];
    f[i + 8] = (_Float16)base[(size_t)(i + 16) * stride];
  }
  return f;
}
__device__ __forceinline__ v16h frag_from_h(const _Float16* __restrict__ p0,
                                            const _Float16* __restrict__ p1) {
  v16h f;
#pragma unroll
  for (int i = 0; i < 8; ++i) { f[i] = p0[i]; f[i + 8] = p1[i]; }
  return f;
}

__device__ __forceinline__ float sigmoidf_(float x) { return 1.f / (1.f + expf(-x)); }

// transpose + convert: src f32 [K][N] row-major -> dst f16 [N][K]
__global__ void cvt_tr_f16_kernel(const float* __restrict__ src, _Float16* __restrict__ dst,
                                  int K, int N)
{
  int idx = blockIdx.x * blockDim.x + threadIdx.x;
  if (idx >= K * N) return;
  int k = idx / N, n = idx - k * N;
  dst[(size_t)n * K + k] = (_Float16)src[idx];
}

// ------------------------------ GEMM kernels --------------------------------
// Weight GEMM: A f32 [M][K] row-major, Bt f16 [N][K] (pre-transposed).
// One wave computes a 16x64 C tile: 4 accumulators share one A-fragment.
__global__ void gemm_rch_kernel(const float* __restrict__ A, int lda,
                                const _Float16* __restrict__ Bt, int ldbt,
                                const float* __restrict__ bias, float bias_scale,
                                float* __restrict__ C, int ldc,
                                int N, int K, float out_scale, int do_relu)
{
  int lane = threadIdx.x & 31;
  int wave = threadIdx.x >> 5;
  int ng = blockIdx.y * 8 + wave;                 // 64-column group
  if (ng * 64 >= N) return;                       // wave-uniform
  int m0 = blockIdx.x << 4, n0 = ng << 6;
  int lo = lane & 15;
  int kh = (lane >> 4) << 3;
  const float* Arow = A + (size_t)(m0 + lo) * lda;
  const _Float16* Bc0 = Bt + (size_t)(n0 + lo) * ldbt;
  v8f acc0 = {}, acc1 = {}, acc2 = {}, acc3 = {};
#pragma unroll 2
  for (int k = 0; k < K; k += 32) {
    __builtin_prefetch(Arow + k + 64, 0, 1);      // global_prefetch_b8
    v16h a = frag_from_f32(Arow + k + kh, Arow + k + kh + 16);
    const _Float16* bp0 = Bc0 + k + kh;
    v16h b0 = frag_from_h(bp0,                bp0 + 16);
    v16h b1 = frag_from_h(bp0 + 16 * ldbt,    bp0 + 16 * ldbt + 16);
    v16h b2 = frag_from_h(bp0 + 32 * ldbt,    bp0 + 32 * ldbt + 16);
    v16h b3 = frag_from_h(bp0 + 48 * ldbt,    bp0 + 48 * ldbt + 16);
    acc0 = __builtin_amdgcn_wmma_f32_16x16x32_f16(false, a, false, b0, (short)0, acc0, false, false);
    acc1 = __builtin_amdgcn_wmma_f32_16x16x32_f16(false, a, false, b1, (short)0, acc1, false, false);
    acc2 = __builtin_amdgcn_wmma_f32_16x16x32_f16(false, a, false, b2, (short)0, acc2, false, false);
    acc3 = __builtin_amdgcn_wmma_f32_16x16x32_f16(false, a, false, b3, (short)0, acc3, false, false);
  }
  int mb = m0 + ((lane >> 4) << 3);
  v8f accs[4] = {acc0, acc1, acc2, acc3};
#pragma unroll
  for (int t = 0; t < 4; ++t) {
    int ncol = n0 + t * 16 + lo;
    float bn = bias ? bias[ncol] * bias_scale : 0.f;
#pragma unroll
    for (int r = 0; r < 8; ++r) {
      float vv = accs[t][r] * out_scale + bn;
      if (do_relu) vv = fmaxf(vv, 0.f);
      C[(size_t)(mb + r) * ldc + ncol] = vv;
    }
  }
}

// Generic batched GEMM, B operand f32 row-major [K][N] (strided gather) + addm.
// batch offset: off(z) = (z/hdiv)*off1 + (z%hdiv)*off2
__global__ void gemm_rr_kernel(const float* __restrict__ A, int lda, int offA1, int offA2,
                               const float* __restrict__ Bm, int ldb, int offB1, int offB2,
                               const float* __restrict__ bias, float bias_scale,
                               const float* __restrict__ addm,
                               float* __restrict__ C, int ldc, int offC1, int offC2,
                               int N, int K, int hdiv, float out_scale, int do_relu)
{
  int z  = blockIdx.z;
  int za = z / hdiv, zb = z - za * hdiv;
  const float* Ab = A + (size_t)za * offA1 + (size_t)zb * offA2;
  const float* Bb = Bm + (size_t)za * offB1 + (size_t)zb * offB2;
  float*       Cb = C + (size_t)za * offC1 + (size_t)zb * offC2;
  const float* Db = addm ? addm + (size_t)za * offC1 + (size_t)zb * offC2 : nullptr;

  int lane = threadIdx.x & 31;
  int wave = threadIdx.x >> 5;
  int nt = blockIdx.y * 8 + wave;
  if (nt * 16 >= N) return;
  int m0 = blockIdx.x << 4, n0 = nt << 4;
  int lo = lane & 15;
  int kh = (lane >> 4) << 3;
  int ncol = n0 + lo;
  const float* Arow = Ab + (size_t)(m0 + lo) * lda;
  v8f acc = {};
  for (int k = 0; k < K; k += 32) {
    __builtin_prefetch(Arow + k + 64, 0, 1);
    v16h a  = frag_from_f32(Arow + k + kh, Arow + k + kh + 16);
    v16h bf = frag_from_f32_strided(Bb + (size_t)(k + kh) * ldb + ncol, ldb);
    acc = __builtin_amdgcn_wmma_f32_16x16x32_f16(false, a, false, bf, (short)0, acc, false, false);
  }
  float bn = bias ? bias[ncol] * bias_scale : 0.f;
  int mb = m0 + ((lane >> 4) << 3);
#pragma unroll
  for (int r = 0; r < 8; ++r) {
    float vv = acc[r] * out_scale + bn;
    if (Db) vv += Db[(size_t)(mb + r) * ldc + ncol];
    if (do_relu) vv = fmaxf(vv, 0.f);
    Cb[(size_t)(mb + r) * ldc + ncol] = vv;
  }
}

// Batched GEMM, B operand f32 column-major Bt[N][K] (contiguous K per lane).
__global__ void gemm_rc_kernel(const float* __restrict__ A, int lda, int offA1, int offA2,
                               const float* __restrict__ Bt, int ldbt, int offB1, int offB2,
                               float* __restrict__ C, int ldc, int offC1, int offC2,
                               int N, int K, int hdiv, float out_scale)
{
  int z  = blockIdx.z;
  int za = z / hdiv, zb = z - za * hdiv;
  const float* Ab = A + (size_t)za * offA1 + (size_t)zb * offA2;
  const float* Bb = Bt + (size_t)za * offB1 + (size_t)zb * offB2;
  float*       Cb = C + (size_t)za * offC1 + (size_t)zb * offC2;

  int lane = threadIdx.x & 31;
  int wave = threadIdx.x >> 5;
  int nt = blockIdx.y * 8 + wave;
  if (nt * 16 >= N) return;
  int m0 = blockIdx.x << 4, n0 = nt << 4;
  int lo = lane & 15;
  int kh = (lane >> 4) << 3;
  int ncol = n0 + lo;
  const float* Arow = Ab + (size_t)(m0 + lo) * lda;
  const float* Bcol = Bb + (size_t)ncol * ldbt;
  v8f acc = {};
  for (int k = 0; k < K; k += 32) {
    __builtin_prefetch(Arow + k + 64, 0, 1);
    v16h a  = frag_from_f32(Arow + k + kh, Arow + k + kh + 16);
    v16h bf = frag_from_f32(Bcol + k + kh, Bcol + k + kh + 16);
    acc = __builtin_amdgcn_wmma_f32_16x16x32_f16(false, a, false, bf, (short)0, acc, false, false);
  }
  int mb = m0 + ((lane >> 4) << 3);
#pragma unroll
  for (int r = 0; r < 8; ++r)
    Cb[(size_t)(mb + r) * ldc + ncol] = acc[r] * out_scale;
}

// ----------------------- fused edge-encoder kernel -------------------------
// ew[b,i,j0+0..15] = sigmoid( mean_n tanh( relu(hi[i]+hj[j]) @ W2[:,n] + b2[n] ) )
// A-panel (16x768, f16) staged in LDS; 6 accumulators per wave (16x96 tile);
// tanh + row-mean fused in epilogue via DS atomics.
__global__ void edge_kernel(const float* __restrict__ hi, const float* __restrict__ hj,
                            const _Float16* __restrict__ w2t, const float* __restrict__ b2,
                            float* __restrict__ ew)
{
  using namespace gcn;
  constexpr int LDA = D + 8;
  __shared__ _Float16 Alds[16 * LDA];
  __shared__ float rowsum[16];
  int jt = blockIdx.x, i = blockIdx.y, b = blockIdx.z;
  int j0 = jt * 16;
  const float* hir = hi + (size_t)(b * S + i) * D;   // hi = x@W1a + b1 (bias folded)
  const float* hjb = hj + (size_t)(b * S + j0) * D;
  if (threadIdx.x < 16) rowsum[threadIdx.x] = 0.f;
  for (int idx = threadIdx.x; idx < 16 * D; idx += 256) {
    int jr = idx / D, d = idx - jr * D;
    float v = hir[d] + hjb[(size_t)jr * D + d];
    Alds[jr * LDA + d] = (_Float16)fmaxf(v, 0.f);
  }
  __syncthreads();

  int lane = threadIdx.x & 31;
  int wave = threadIdx.x >> 5;
  int lo = lane & 15;
  int kh = (lane >> 4) << 3;
  const _Float16* Bc = w2t + (size_t)(wave * 96 + lo) * D;
  v8f a0 = {}, a1 = {}, a2 = {}, a3 = {}, a4 = {}, a5 = {};
#pragma unroll 2
  for (int k = 0; k < D; k += 32) {
    const _Float16* ap = &Alds[lo * LDA + k + kh];
    v16h a = frag_from_h(ap, ap + 16);
    const _Float16* bp = Bc + k + kh;
    v16h b0 = frag_from_h(bp,             bp + 16);
    v16h b1 = frag_from_h(bp + 16 * D,    bp + 16 * D + 16);
    v16h b2f = frag_from_h(bp + 32 * D,   bp + 32 * D + 16);
    v16h b3 = frag_from_h(bp + 48 * D,    bp + 48 * D + 16);
    v16h b4 = frag_from_h(bp + 64 * D,    bp + 64 * D + 16);
    v16h b5 = frag_from_h(bp + 80 * D,    bp + 80 * D + 16);
    a0 = __builtin_amdgcn_wmma_f32_16x16x32_f16(false, a, false, b0,  (short)0, a0, false, false);
    a1 = __builtin_amdgcn_wmma_f32_16x16x32_f16(false, a, false, b1,  (short)0, a1, false, false);
    a2 = __builtin_amdgcn_wmma_f32_16x16x32_f16(false, a, false, b2f, (short)0, a2, false, false);
    a3 = __builtin_amdgcn_wmma_f32_16x16x32_f16(false, a, false, b3,  (short)0, a3, false, false);
    a4 = __builtin_amdgcn_wmma_f32_16x16x32_f16(false, a, false, b4,  (short)0, a4, false, false);
    a5 = __builtin_amdgcn_wmma_f32_16x16x32_f16(false, a, false, b5,  (short)0, a5, false, false);
  }
  float psum[8];
#pragma unroll
  for (int r = 0; r < 8; ++r) psum[r] = 0.f;
  v8f accs[6] = {a0, a1, a2, a3, a4, a5};
#pragma unroll
  for (int t = 0; t < 6; ++t) {
    float bn = b2[wave * 96 + t * 16 + lo];
#pragma unroll
    for (int r = 0; r < 8; ++r) psum[r] += tanhf(accs[t][r] + bn);
  }
#pragma unroll
  for (int r = 0; r < 8; ++r) {
    int mrow = r + ((lane >> 4) << 3);
    atomicAdd(&rowsum[mrow], psum[r]);
  }
  __syncthreads();
  if (threadIdx.x < 16) {
    float m = rowsum[threadIdx.x] * (1.f / (float)D);
    ew[(size_t)(b * S + i) * S + j0 + threadIdx.x] = sigmoidf_(m);
  }
}

// ------------------------- small support kernels ---------------------------
__global__ void adj_softmax_kernel(const float* __restrict__ adj, const float* __restrict__ ew,
                                   float* __restrict__ adj_e, float* __restrict__ denom)
{
  using namespace gcn;
  __shared__ float red[S];
  int i = blockIdx.x, b = blockIdx.y, j = threadIdx.x;
  size_t base = (size_t)(b * S + i) * S;
  float e = adj[base + j] * ew[base + j] + ((j == i) ? 1.f : 0.f);
  red[j] = e; __syncthreads();
  for (int off = S / 2; off > 0; off >>= 1) {
    if (j < off) red[j] = fmaxf(red[j], red[j + off]);
    __syncthreads();
  }
  float mx = red[0]; __syncthreads();
  float ex = expf(e - mx);
  red[j] = ex; __syncthreads();
  for (int off = S / 2; off > 0; off >>= 1) {
    if (j < off) red[j] += red[j + off];
    __syncthreads();
  }
  float sm = ex / red[0];
  adj_e[base + j] = sm;
  __syncthreads();
  red[j] = sm; __syncthreads();
  for (int off = S / 2; off > 0; off >>= 1) {
    if (j < off) red[j] += red[j + off];
    __syncthreads();
  }
  if (j == 0) denom[b * S + i] = red[0] + 1.f;
}

__global__ void colsum_kernel(const float* __restrict__ adj_e, float* __restrict__ colsum)
{
  using namespace gcn;
  int idx = blockIdx.x * blockDim.x + threadIdx.x;
  if (idx >= MR) return;
  int b = idx / S, j = idx - b * S;
  float s = 0.f;
  for (int i = 0; i < S; ++i) s += adj_e[(size_t)(b * S + i) * S + j];
  colsum[idx] = s;
}

__global__ void mask_kernel(const float* __restrict__ denom, const float* __restrict__ colsum,
                            float* __restrict__ mask_out)
{
  using namespace gcn;
  int idx = blockIdx.x * blockDim.x + threadIdx.x;
  if (idx >= MR) return;
  float rs = denom[idx] - 1.f;
  mask_out[idx] = ((rs + colsum[idx]) == 0.f) ? 1.f : 0.f;
}

__global__ void conv1d_kernel(const float* __restrict__ x, const float* __restrict__ w,
                              const float* __restrict__ bias, float* __restrict__ y,
                              int Kk, int pad)
{
  using namespace gcn;
  int idx = blockIdx.x * blockDim.x + threadIdx.x;
  if (idx >= Bn * D * LC) return;
  int b = idx / (D * LC);
  int rem = idx - b * (D * LC);
  int o = rem / LC, t = rem - o * LC;
  float acc = bias[o];
  for (int kk = 0; kk < Kk; ++kk) {
    int sp = t - pad + kk;
    if (sp < 0 || sp >= S) continue;
    const float* xr = x + (size_t)(b * S + sp) * D;
    const float* wr = w + (size_t)o * D * Kk + kk;
    for (int c = 0; c < D; ++c) acc += xr[c] * wr[(size_t)c * Kk];
  }
  y[idx] = fmaxf(acc, 0.f);
}

__global__ void interp_kernel(const float* __restrict__ y129, float* __restrict__ msI,
                              int chan_off)
{
  using namespace gcn;
  int idx = blockIdx.x * blockDim.x + threadIdx.x;
  if (idx >= Bn * D * S) return;
  int b = idx / (D * S);
  int rem = idx - b * (D * S);
  int o = rem / S, t = rem - o * S;
  float src = ((float)t + 0.5f) * ((float)LC / (float)S) - 0.5f;
  src = fmaxf(src, 0.f);
  int i0 = (int)floorf(src);
  if (i0 > LC - 1) i0 = LC - 1;
  int i1 = i0 + 1; if (i1 > LC - 1) i1 = LC - 1;
  float lam = src - (float)i0;
  const float* row = y129 + (size_t)(b * D + o) * LC;
  float v = row[i0] * (1.f - lam) + row[i1] * lam;
  msI[((size_t)b * D3 + chan_off + o) * S + t] = v;
}

__global__ void attn_softmax_kernel(float* __restrict__ sc, const float* __restrict__ adj_e)
{
  using namespace gcn;
  __shared__ float red[S];
  int m = blockIdx.x, z = blockIdx.y, j = threadIdx.x;
  int b = z / H;
  size_t base = (size_t)z * SS + (size_t)m * S;
  float v = sc[base + j];
  if (adj_e[(size_t)(b * S + m) * S + j] == 0.f) v = -1e9f;
  red[j] = v; __syncthreads();
  for (int off = S / 2; off > 0; off >>= 1) {
    if (j < off) red[j] = fmaxf(red[j], red[j + off]);
    __syncthreads();
  }
  float mx = red[0]; __syncthreads();
  float ex = expf(v - mx);
  red[j] = ex; __syncthreads();
  for (int off = S / 2; off > 0; off >>= 1) {
    if (j < off) red[j] += red[j + off];
    __syncthreads();
  }
  sc[base + j] = ex / red[0];
}

__global__ void ln_kernel(const float* __restrict__ x, const float* __restrict__ resid,
                          const float* __restrict__ g, const float* __restrict__ bta,
                          float* __restrict__ y, int Dn)
{
  __shared__ float red[256];
  int row = blockIdx.x, tid = threadIdx.x;
  const float* xr = x + (size_t)row * Dn;
  const float* rr = resid ? resid + (size_t)row * Dn : nullptr;
  float s = 0.f;
  for (int i = tid; i < Dn; i += 256) s += xr[i] + (rr ? rr[i] : 0.f);
  red[tid] = s; __syncthreads();
  for (int off = 128; off > 0; off >>= 1) {
    if (tid < off) red[tid] += red[tid + off];
    __syncthreads();
  }
  float mean = red[0] / (float)Dn; __syncthreads();
  float s2 = 0.f;
  for (int i = tid; i < Dn; i += 256) {
    float v = xr[i] + (rr ? rr[i] : 0.f) - mean;
    s2 += v * v;
  }
  red[tid] = s2; __syncthreads();
  for (int off = 128; off > 0; off >>= 1) {
    if (tid < off) red[tid] += red[tid + off];
    __syncthreads();
  }
  float inv = rsqrtf(red[0] / (float)Dn + 1e-5f);
  for (int i = tid; i < Dn; i += 256) {
    float v = xr[i] + (rr ? rr[i] : 0.f);
    y[(size_t)row * Dn + i] = (v - mean) * inv * g[i] + bta[i];
  }
}

__global__ void mix_kernel(const float* __restrict__ rawAxW, const float* __restrict__ denom,
                           const float* __restrict__ attLN, const float* __restrict__ attn_w,
                           const float* __restrict__ res_w, int l,
                           const float* __restrict__ res, float* __restrict__ preln)
{
  using namespace gcn;
  int idx = blockIdx.x * blockDim.x + threadIdx.x;
  if (idx >= MR * D) return;
  int row = idx / D;
  float aw = sigmoidf_(attn_w[l]);
  float z = aw * (rawAxW[idx] / denom[row]) + (1.f - aw) * attLN[idx];
  if (res) {
    float rw = sigmoidf_(res_w[l]);
    z = rw * z + (1.f - rw) * res[idx];
  }
  preln[idx] = fmaxf(z, 0.f);
}

__global__ void concat_kernel(const float* __restrict__ a, const float* __restrict__ b,
                              float* __restrict__ cat)
{
  using namespace gcn;
  int idx = blockIdx.x * blockDim.x + threadIdx.x;
  if (idx >= MR * 2 * D) return;
  int row = idx / (2 * D), c = idx - row * (2 * D);
  cat[idx] = (c < D) ? a[(size_t)row * D + c] : b[(size_t)row * D + (c - D)];
}

// ------------------------------- host side ---------------------------------
static inline void cvt_tr(hipStream_t st, const float* src, _Float16* dst, int K, int N)
{
  cvt_tr_f16_kernel<<<(K * N + 255) / 256, 256, 0, st>>>(src, dst, K, N);
}
static inline void gemm_w(hipStream_t st, const float* A, int lda, const _Float16* Bt, int ldbt,
                          const float* bias, float bscale, float* C, int ldc,
                          int M, int N, int K, float oscale, int relu)
{
  dim3 grid(M / 16, (N / 64 + 7) / 8, 1);
  gemm_rch_kernel<<<grid, 256, 0, st>>>(A, lda, Bt, ldbt, bias, bscale, C, ldc,
                                        N, K, oscale, relu);
}
static inline void gemm_rr(hipStream_t st, const float* A, int lda, int oA1, int oA2,
                           const float* Bm, int ldb, int oB1, int oB2,
                           const float* bias, float bscale, const float* addm,
                           float* C, int ldc, int oC1, int oC2,
                           int M, int N, int K, int Z, int hdiv, float oscale, int relu)
{
  dim3 grid(M / 16, (N / 16 + 7) / 8, Z);
  gemm_rr_kernel<<<grid, 256, 0, st>>>(A, lda, oA1, oA2, Bm, ldb, oB1, oB2,
                                       bias, bscale, addm, C, ldc, oC1, oC2,
                                       N, K, hdiv, oscale, relu);
}
static inline void gemm_rc(hipStream_t st, const float* A, int lda, int oA1, int oA2,
                           const float* Bt, int ldbt, int oB1, int oB2,
                           float* C, int ldc, int oC1, int oC2,
                           int M, int N, int K, int Z, int hdiv, float oscale)
{
  dim3 grid(M / 16, (N / 16 + 7) / 8, Z);
  gemm_rc_kernel<<<grid, 256, 0, st>>>(A, lda, oA1, oA2, Bt, ldbt, oB1, oB2,
                                       C, ldc, oC1, oC2, N, K, hdiv, oscale);
}

extern "C" void kernel_launch(void* const* d_in, const int* in_sizes, int n_in,
                              void* d_out, int out_size, void* d_ws, size_t ws_size,
                              hipStream_t stream)
{
  using namespace gcn;
  (void)in_sizes; (void)n_in; (void)out_size; (void)ws_size;
  const float* adj     = (const float*)d_in[0];
  const float* x       = (const float*)d_in[1];
  const float* edge_w1 = (const float*)d_in[2];
  const float* edge_b1 = (const float*)d_in[3];
  const float* edge_w2 = (const float*)d_in[4];
  const float* edge_b2 = (const float*)d_in[5];
  const float* W_w     = (const float*)d_in[6];
  const float* W_b     = (const float*)d_in[7];
  const float* ln_g    = (const float*)d_in[8];
  const float* ln_b    = (const float*)d_in[9];
  const float* res_w   = (const float*)d_in[10];
  const float* attn_w  = (const float*)d_in[11];
  const float* q_w = (const float*)d_in[12]; const float* q_b = (const float*)d_in[13];
  const float* k_w = (const float*)d_in[14]; const float* k_b = (const float*)d_in[15];
  const float* v_w = (const float*)d_in[16]; const float* v_b = (const float*)d_in[17];
  const float* o_w = (const float*)d_in[18]; const float* o_b = (const float*)d_in[19];
  const float* attn_ln_g = (const float*)d_in[20]; const float* attn_ln_b = (const float*)d_in[21];
  const float* fus_w1 = (const float*)d_in[22]; const float* fus_b1 = (const float*)d_in[23];
  const float* fus_w2 = (const float*)d_in[24]; const float* fus_b2 = (const float*)d_in[25];
  const float* fe_w1  = (const float*)d_in[26]; const float* fe_b1  = (const float*)d_in[27];
  const float* fe_w2  = (const float*)d_in[28]; const float* fe_b2  = (const float*)d_in[29];
  const float* fe_ln_g = (const float*)d_in[30]; const float* fe_ln_b = (const float*)d_in[31];
  const float* ms_w[3] = {(const float*)d_in[32], (const float*)d_in[34], (const float*)d_in[36]};
  const float* ms_b[3] = {(const float*)d_in[33], (const float*)d_in[35], (const float*)d_in[37]};

  float* ws = (float*)d_ws;
  float* hi    = ws + W_HI;   float* hj    = ws + W_HJ;
  float* ew    = ws + W_EW;   float* adj_e = ws + W_ADJE;
  float* denom = ws + W_DEN;  float* colsum = ws + W_COL;
  float* c129  = ws + W_C129; float* msI   = ws + W_MSI;
  float* g     = ws + W_G;    float* fush  = ws + W_FUSH;
  float* msf   = ws + W_MSF;
  float* qb = ws + W_Q; float* kb = ws + W_K; float* vb = ws + W_V;
  float* attb = ws + W_ATT;  float* ctx  = ws + W_CTX;
  float* atto = ws + W_ATTO; float* attLN = ws + W_ATTLN;
  float* tbuf = ws + W_TBUF; float* rawAxW = ws + W_RAXW;
  float* preln = ws + W_PRELN;
  float* outA = ws + W_OUTA; float* outB = ws + W_OUTB;
  float* cat  = ws + W_CAT;  float* feh  = ws + W_FEH;  float* fe2 = ws + W_FE2;
  _Float16* wt = (_Float16*)(ws + W_END);   // f16 transposed weight region
  float* out_final = (float*)d_out;
  float* out_mask  = out_final + (size_t)MR * D;

  const float inv_sqrt_dh = 0.1020620726159658f;  // 1/sqrt(96)

  // ---- weight convert + transpose to f16 [N][K] (L2-resident, ~20MB) ------
  cvt_tr(stream, edge_w1,            wt + F_EW1A, D, D);
  cvt_tr(stream, edge_w1 + (size_t)DD, wt + F_EW1B, D, D);
  cvt_tr(stream, edge_w2,            wt + F_EW2,  D, D);
  for (int l = 0; l < NL; ++l)
    cvt_tr(stream, W_w + (size_t)l * DD, wt + F_WW + (size_t)l * DD, D, D);
  cvt_tr(stream, q_w,    wt + F_QW,   D, D);
  cvt_tr(stream, k_w,    wt + F_KW,   D, D);
  cvt_tr(stream, v_w,    wt + F_VW,   D, D);
  cvt_tr(stream, o_w,    wt + F_OW,   D, D);
  cvt_tr(stream, fus_w1, wt + F_FUS1, D3, D);
  cvt_tr(stream, fus_w2, wt + F_FUS2, D, D);
  cvt_tr(stream, fe_w1,  wt + F_FE1,  2 * D, D);
  cvt_tr(stream, fe_w2,  wt + F_FE2,  D, D);

  // ---- edge encoder -------------------------------------------------------
  gemm_w(stream, x, D, wt + F_EW1A, D, edge_b1, 1.f, hi, D, MR, D, D, 1.f, 0);
  gemm_w(stream, x, D, wt + F_EW1B, D, nullptr, 0.f, hj, D, MR, D, D, 1.f, 0);
  edge_kernel<<<dim3(S / 16, S, Bn), 256, 0, stream>>>(hi, hj, wt + F_EW2, edge_b2, ew);
  adj_softmax_kernel<<<dim3(S, Bn), S, 0, stream>>>(adj, ew, adj_e, denom);
  colsum_kernel<<<2, 256, 0, stream>>>(adj_e, colsum);

  // ---- multi-scale --------------------------------------------------------
  for (int sc = 0; sc < 3; ++sc) {
    int Kk = 2 << sc, pad = 1 << sc;
    conv1d_kernel<<<(Bn * D * LC + 255) / 256, 256, 0, stream>>>(x, ms_w[sc], ms_b[sc],
                                                                 c129, Kk, pad);
    interp_kernel<<<(Bn * D * S + 255) / 256, 256, 0, stream>>>(c129, msI, sc * D);
  }
  gemm_rc(stream, adj_e, S, SS, 0, msI, S, D3 * S, 0,
          g, D3, S * D3, 0, S, D3, S, Bn, 1, 1.f);
  gemm_w(stream, g, D3, wt + F_FUS1, D3, fus_b1, 1.f, fush, D, MR, D, D3, 1.f, 1);
  gemm_w(stream, fush, D, wt + F_FUS2, D, fus_b2, 1.f, msf, D, MR, D, D, 1.f, 0);

  // ---- GCN layers ---------------------------------------------------------
  for (int l = 0; l < NL; ++l) {
    const float* in = (l == 0) ? x : ((l == 1) ? outA : outB);
    float* outdst   = (l == 1) ? outB : outA;
    const float* resv = (l == 0) ? nullptr : ((l == 1) ? outA : outB);

    // tbuf = adj_e @ in + in
    gemm_rr(stream, adj_e, S, SS, 0, in, D, SD, 0, nullptr, 0.f, in,
            tbuf, D, SD, 0, S, D, S, Bn, 1, 1.f, 0);
    // rawAxW = tbuf @ W[l] + 2*b[l]
    gemm_w(stream, tbuf, D, wt + F_WW + (size_t)l * DD, D,
           W_b + (size_t)l * D, 2.f, rawAxW, D, MR, D, D, 1.f, 0);

    // attention projections
    gemm_w(stream, in, D, wt + F_QW, D, q_b, 1.f, qb, D, MR, D, D, 1.f, 0);
    gemm_w(stream, in, D, wt + F_KW, D, k_b, 1.f, kb, D, MR, D, D, 1.f, 0);
    gemm_w(stream, in, D, wt + F_VW, D, v_b, 1.f, vb, D, MR, D, D, 1.f, 0);
    // scores[z=b*H+h] = q_h @ k_h^T / sqrt(dh)
    gemm_rc(stream, qb, D, SD, DH, kb, D, SD, DH,
            attb, S, H * SS, SS, S, S, DH, Bn * H, H, inv_sqrt_dh);
    attn_softmax_kernel<<<dim3(S, Bn * H), S, 0, stream>>>(attb, adj_e);
    // ctx = attn @ v_h
    gemm_rr(stream, attb, S, H * SS, SS, vb, D, SD, DH, nullptr, 0.f, nullptr,
            ctx, D, SD, DH, S, DH, S, Bn * H, H, 1.f, 0);
    gemm_w(stream, ctx, D, wt + F_OW, D, o_b, 1.f, atto, D, MR, D, D, 1.f, 0);
    ln_kernel<<<MR, 256, 0, stream>>>(atto, in, attn_ln_g, attn_ln_b, attLN, D);

    mix_kernel<<<(MR * D + 255) / 256, 256, 0, stream>>>(rawAxW, denom, attLN,
                                                         attn_w, res_w, l, resv, preln);
    ln_kernel<<<MR, 256, 0, stream>>>(preln, nullptr, ln_g + (size_t)l * D,
                                      ln_b + (size_t)l * D, outdst, D);
  }

  // ---- feature extractor --------------------------------------------------
  concat_kernel<<<(MR * 2 * D + 255) / 256, 256, 0, stream>>>(outA, msf, cat);
  gemm_w(stream, cat, 2 * D, wt + F_FE1, 2 * D, fe_b1, 1.f, feh, D, MR, D, 2 * D, 1.f, 1);
  gemm_w(stream, feh, D, wt + F_FE2, D, fe_b2, 1.f, fe2, D, MR, D, D, 1.f, 0);
  ln_kernel<<<MR, 256, 0, stream>>>(fe2, nullptr, fe_ln_g, fe_ln_b, out_final, D);
  mask_kernel<<<2, 256, 0, stream>>>(denom, colsum, out_mask);
}
